// RainbowEvolution_80633716015694
// MI455X (gfx1250) — matrix operand order repaired and checked by reading, not compiled, not run on hardware
//
#include <hip/hip_runtime.h>
#include <math.h>

// Problem sizes (fixed by the reference)
#define T_ 1024
#define L_ 32
#define D_ 512
#define P_ 512
#define H_ 2048

typedef __attribute__((ext_vector_type(16))) _Float16 v16h;
typedef __attribute__((ext_vector_type(8)))  float    v8f;
typedef __attribute__((ext_vector_type(4)))  float    f32x4;   // native vec for NT store

// ---------------------------------------------------------------------------
// bmean[t,d] = mean_l base[t,l,d]  (float4 pass over 67MB; RT hint keeps the
// tensor L2-resident for the later weighted-T reduction — L2 is 192MB)
// ---------------------------------------------------------------------------
__global__ void k_bmean(const float* __restrict__ base, float* __restrict__ bmean) {
  int t = blockIdx.x, d4 = threadIdx.x;                 // 128 threads * float4
  const float4* p = (const float4*)(base + (size_t)t * (L_ * D_)) + d4;
  float4 s = make_float4(0.f, 0.f, 0.f, 0.f);
#pragma unroll
  for (int l = 0; l < L_; ++l) {
    float4 v = p[l * (D_ / 4)];
    s.x += v.x; s.y += v.y; s.z += v.z; s.w += v.w;
  }
  const float r = 1.0f / L_;
  ((float4*)(bmean + (size_t)t * D_))[d4] = make_float4(s.x * r, s.y * r, s.z * r, s.w * r);
}

// task_vec = tanh(task_embedding @ Wt + bt)   (M=1 GEMV, trivial)
__global__ void k_taskvec(const float* __restrict__ e, const float* __restrict__ Wt,
                          const float* __restrict__ bt, float* __restrict__ tv) {
  int p = threadIdx.x;
  float s = bt[p];
  for (int d = 0; d < D_; ++d) s += e[d] * Wt[d * P_ + p];
  tv[p] = tanhf(s);
}

// ---------------------------------------------------------------------------
// WMMA GEMM: C[M,N] = act(A[M,K] @ B[K,N] + bias[N])
// One wave computes a 16x64 strip: 4 accumulators share one A fragment,
// 4 independent v_wmma_f32_16x16x32_f16 per K-step. A fragment loads are
// b128 (layout: e<8 -> K=k0+8g+e, e>=8 -> K=k0+16+8g+(e-8), both contiguous).
// ACT: 0=none, 1=tanh, 2=relu
// ---------------------------------------------------------------------------
template <int ACT>
__global__ void k_wmma_gemm(const float* __restrict__ A, const float* __restrict__ B,
                            const float* __restrict__ bias, float* __restrict__ C,
                            int M, int N, int K) {
  int wave = blockIdx.x * (blockDim.x >> 5) + (threadIdx.x >> 5);
  int lane = threadIdx.x & 31;
  int tilesN = N >> 6;                        // 64-wide strips
  int tiles  = (M >> 4) * tilesN;
  if (wave >= tiles) return;                  // wave-uniform: EXEC stays all-ones
  int tm = wave / tilesN, tn = wave - tm * tilesN;
  int g = lane >> 4, ln = lane & 15;

  v8f acc[4] = {};
  const float* Arow  = A + (size_t)(tm * 16 + ln) * K;
  const float* Bbase = B + (tn * 64 + ln);

  for (int k0 = 0; k0 < K; k0 += 32) {
    // --- A fragment: 4x b128 loads, cvt to f16 ---
    const float* Ap = Arow + k0 + 8 * g;
    float4 x0 = *(const float4*)(Ap);
    float4 x1 = *(const float4*)(Ap + 4);
    float4 x2 = *(const float4*)(Ap + 16);
    float4 x3 = *(const float4*)(Ap + 20);
    v16h a;
    a[0] = (_Float16)x0.x; a[1] = (_Float16)x0.y; a[2]  = (_Float16)x0.z; a[3]  = (_Float16)x0.w;
    a[4] = (_Float16)x1.x; a[5] = (_Float16)x1.y; a[6]  = (_Float16)x1.z; a[7]  = (_Float16)x1.w;
    a[8] = (_Float16)x2.x; a[9] = (_Float16)x2.y; a[10] = (_Float16)x2.z; a[11] = (_Float16)x2.w;
    a[12] = (_Float16)x3.x; a[13] = (_Float16)x3.y; a[14] = (_Float16)x3.z; a[15] = (_Float16)x3.w;

    // --- B fragments for 4 column tiles (coalesced across lanes) ---
    const float* Bp = Bbase + (size_t)(k0 + 8 * g) * N;
    v16h b0, b1, b2, b3;
#pragma unroll
    for (int e = 0; e < 16; ++e) {
      const float* r = Bp + (size_t)(e + (e & 8)) * N;
      b0[e] = (_Float16)r[0];
      b1[e] = (_Float16)r[16];
      b2[e] = (_Float16)r[32];
      b3[e] = (_Float16)r[48];
    }

    acc[0] = __builtin_amdgcn_wmma_f32_16x16x32_f16(false, a, false, b0, (short)0, acc[0], false, false);
    acc[1] = __builtin_amdgcn_wmma_f32_16x16x32_f16(false, a, false, b1, (short)0, acc[1], false, false);
    acc[2] = __builtin_amdgcn_wmma_f32_16x16x32_f16(false, a, false, b2, (short)0, acc[2], false, false);
    acc[3] = __builtin_amdgcn_wmma_f32_16x16x32_f16(false, a, false, b3, (short)0, acc[3], false, false);
  }

#pragma unroll
  for (int q = 0; q < 4; ++q) {
    int col = tn * 64 + 16 * q + ln;
    float bv = bias[col];
#pragma unroll
    for (int r = 0; r < 8; ++r) {
      float v = acc[q][r] + bv;
      if (ACT == 1) v = tanhf(v);
      if (ACT == 2) v = fmaxf(v, 0.0f);
      C[(size_t)(tm * 16 + 8 * g + r) * N + col] = v;
    }
  }
}

// scores[t] = scale * dot(key_repr[t,:], task_vec)
__global__ void k_scores(const float* __restrict__ kr, const float* __restrict__ tv,
                         float* __restrict__ sc, float scale) {
  __shared__ float red[256];
  int t = blockIdx.x, tid = threadIdx.x;
  float s = 0.0f;
  for (int p = tid; p < P_; p += 256) s += kr[t * P_ + p] * tv[p];
  red[tid] = s; __syncthreads();
  for (int o = 128; o > 0; o >>= 1) { if (tid < o) red[tid] += red[tid + o]; __syncthreads(); }
  if (tid == 0) sc[t] = red[0] * scale;
}

// softmax over exactly 1024 elements, single block
__global__ void k_softmax(const float* __restrict__ x, float* __restrict__ y) {
  __shared__ float red[1024];
  int t = threadIdx.x;
  float v = x[t];
  red[t] = v; __syncthreads();
  for (int o = 512; o > 0; o >>= 1) { if (t < o) red[t] = fmaxf(red[t], red[t + o]); __syncthreads(); }
  float mx = red[0]; __syncthreads();
  float e = __expf(v - mx);
  red[t] = e; __syncthreads();
  for (int o = 512; o > 0; o >>= 1) { if (t < o) red[t] += red[t + o]; __syncthreads(); }
  y[t] = e / red[0];
}

// pooled_q[p] = mean_l query[l,p]
__global__ void k_pooledq(const float* __restrict__ q, float* __restrict__ pq) {
  int p = threadIdx.x;
  float s = 0.0f;
#pragma unroll
  for (int l = 0; l < L_; ++l) s += q[l * P_ + p];
  pq[p] = s * (1.0f / L_);
}

// s2[t] = scale * dot( (1+w0[t])*(key_repr[t]-bk)+bk , pooled_q )
__global__ void k_s2(const float* __restrict__ kr, const float* __restrict__ bk,
                     const float* __restrict__ pq, const float* __restrict__ w0,
                     float* __restrict__ s2, float scale) {
  __shared__ float red[256];
  int t = blockIdx.x, tid = threadIdx.x;
  float wm = 1.0f + w0[t];
  float s = 0.0f;
  for (int p = tid; p < P_; p += 256) {
    float pk = wm * (kr[t * P_ + p] - bk[p]) + bk[p];
    s += pk * pq[p];
  }
  red[tid] = s; __syncthreads();
  for (int o = 128; o > 0; o >>= 1) { if (tid < o) red[tid] += red[tid + o]; __syncthreads(); }
  if (tid == 0) s2[t] = red[0] * scale;
}

// ta = softmax(s2); w = normalize(0.5*w0 + 0.5*ta); c = w*(1+w0); emit task_weights
__global__ void k_combine(const float* __restrict__ s2, const float* __restrict__ w0,
                          float* __restrict__ w, float* __restrict__ c,
                          float* __restrict__ out_tw) {
  __shared__ float red[1024];
  int t = threadIdx.x;
  float v = s2[t];
  red[t] = v; __syncthreads();
  for (int o = 512; o > 0; o >>= 1) { if (t < o) red[t] = fmaxf(red[t], red[t + o]); __syncthreads(); }
  float mx = red[0]; __syncthreads();
  float e = __expf(v - mx);
  red[t] = e; __syncthreads();
  for (int o = 512; o > 0; o >>= 1) { if (t < o) red[t] += red[t + o]; __syncthreads(); }
  float ta = e / red[0]; __syncthreads();
  float wi = 0.5f * w0[t] + 0.5f * ta;
  red[t] = wi; __syncthreads();
  for (int o = 512; o > 0; o >>= 1) { if (t < o) red[t] += red[t + o]; __syncthreads(); }
  wi = wi / fmaxf(red[0], 1e-6f);
  w[t] = wi;
  c[t] = wi * (1.0f + w0[t]);
  out_tw[t] = wi;
}

// weighted_prompts = sum_t c[t]*base[t,:,:] — two-stage deterministic, float4.
// Second full-tensor read: likely an L2 hit (base stayed resident after k_bmean).
__global__ void k_wp_partial(const float* __restrict__ base, const float* __restrict__ c,
                             float* __restrict__ partial) {
  int idx = blockIdx.x * 256 + threadIdx.x;   // 0..4095 (float4 index into [L,D])
  int tc  = blockIdx.y;                       // 32 chunks of 32 tasks
  const float4* b4 = (const float4*)base;
  float4 s = make_float4(0.f, 0.f, 0.f, 0.f);
  int t0 = tc * 32;
#pragma unroll 4
  for (int t = t0; t < t0 + 32; ++t) {
    float ct = c[t];
    float4 v = b4[(size_t)t * (L_ * D_ / 4) + idx];
    s.x += ct * v.x; s.y += ct * v.y; s.z += ct * v.z; s.w += ct * v.w;
  }
  ((float4*)partial)[(size_t)tc * (L_ * D_ / 4) + idx] = s;
}

__global__ void k_wp_reduce(const float* __restrict__ partial, float* __restrict__ wp) {
  int idx = blockIdx.x * 256 + threadIdx.x;
  const float4* p4 = (const float4*)partial;
  float4 s = make_float4(0.f, 0.f, 0.f, 0.f);
#pragma unroll
  for (int j = 0; j < 32; ++j) {
    float4 v = p4[(size_t)j * (L_ * D_ / 4) + idx];
    s.x += v.x; s.y += v.y; s.z += v.z; s.w += v.w;
  }
  ((float4*)wp)[idx] = s;
}

// feature attention (32x32) + evolved_proj = attn @ weighted_values; emits feature_attn
__global__ void k_feature(const float* __restrict__ q, const float* __restrict__ wk,
                          const float* __restrict__ wv, float* __restrict__ ep,
                          float* __restrict__ out_attn, float scale) {
  __shared__ float attn[L_][L_];
  int tid = threadIdx.x;           // 1024 threads
  int i = tid >> 5, j = tid & 31;
  float s = 0.0f;
  for (int p = 0; p < P_; ++p) s += q[i * P_ + p] * wk[j * P_ + p];
  attn[i][j] = s * scale;
  __syncthreads();
  if (tid < L_) {
    float mx = -1e30f;
    for (int jj = 0; jj < L_; ++jj) mx = fmaxf(mx, attn[tid][jj]);
    float sum = 0.0f;
    for (int jj = 0; jj < L_; ++jj) { float e = __expf(attn[tid][jj] - mx); attn[tid][jj] = e; sum += e; }
    float inv = 1.0f / sum;
    for (int jj = 0; jj < L_; ++jj) attn[tid][jj] *= inv;
  }
  __syncthreads();
  out_attn[tid] = attn[i][j];
  for (int o = tid; o < L_ * P_; o += 1024) {
    int l = o >> 9, p = o & (P_ - 1);
    float acc = 0.0f;
#pragma unroll
    for (int jj = 0; jj < L_; ++jj) acc += attn[l][jj] * wv[jj * P_ + p];
    ep[o] = acc;
  }
}

// out[row,:] = LayerNorm(x[row,:] + y[row,:]) * g + b   (D=512, one block/row)
__global__ void k_addln(const float* __restrict__ x, const float* __restrict__ y,
                        const float* __restrict__ gw, const float* __restrict__ bw,
                        float* __restrict__ out) {
  __shared__ float red[D_];
  int row = blockIdx.x, tid = threadIdx.x;
  float v = x[row * D_ + tid] + y[row * D_ + tid];
  red[tid] = v; __syncthreads();
  for (int o = D_ / 2; o > 0; o >>= 1) { if (tid < o) red[tid] += red[tid + o]; __syncthreads(); }
  float m = red[0] * (1.0f / D_); __syncthreads();
  float d = v - m;
  red[tid] = d * d; __syncthreads();
  for (int o = D_ / 2; o > 0; o >>= 1) { if (tid < o) red[tid] += red[tid + o]; __syncthreads(); }
  float var = red[0] * (1.0f / D_);
  out[row * D_ + tid] = d * rsqrtf(var + 1e-5f) * gw[tid] + bw[tid];
}

// aligned_prompts[t,:,:] = aligned — 67MB streaming write, b128 non-temporal
// stores (native ext-vector type) so the output stream doesn't evict
// base_prompts / the source tile from L2.
__global__ void k_broadcast(const f32x4* __restrict__ al, f32x4* __restrict__ out) {
  size_t gid = (size_t)blockIdx.x * 256 + threadIdx.x;
  f32x4 v = al[gid & (size_t)(L_ * D_ / 4 - 1)];
  __builtin_nontemporal_store(v, out + gid);
}

// ---------------------------------------------------------------------------
extern "C" void kernel_launch(void* const* d_in, const int* in_sizes, int n_in,
                              void* d_out, int out_size, void* d_ws, size_t ws_size,
                              hipStream_t stream) {
  (void)in_sizes; (void)n_in; (void)out_size; (void)ws_size;
  const float* base = (const float*)d_in[0];
  const float* newp = (const float*)d_in[1];
  const float* temb = (const float*)d_in[2];
  const float* Wt  = (const float*)d_in[3];  const float* bt  = (const float*)d_in[4];
  const float* Wq  = (const float*)d_in[5];  const float* bq  = (const float*)d_in[6];
  const float* Wk  = (const float*)d_in[7];  const float* bk  = (const float*)d_in[8];
  const float* Wv  = (const float*)d_in[9];  const float* bv  = (const float*)d_in[10];
  const float* Wo  = (const float*)d_in[11]; const float* bo  = (const float*)d_in[12];
  const float* Wa1 = (const float*)d_in[13]; const float* ba1 = (const float*)d_in[14];
  const float* Wa2 = (const float*)d_in[15]; const float* ba2 = (const float*)d_in[16];
  const float* g_in  = (const float*)d_in[17]; const float* b_in  = (const float*)d_in[18];
  const float* g_out = (const float*)d_in[19]; const float* b_out = (const float*)d_in[20];

  // workspace layout (floats); partials alias bmean (dead after key_repr GEMM)
  float* ws       = (float*)d_ws;
  float* bmean    = ws;                     // 524288
  float* key_repr = bmean + T_ * D_;        // 524288
  float* task_vec = key_repr + T_ * P_;     // 512
  float* query    = task_vec + P_;          // 16384
  float* scores   = query + L_ * P_;        // 1024
  float* w0       = scores + T_;            // 1024
  float* pooled_q = w0 + T_;                // 512
  float* s2       = pooled_q + P_;          // 1024
  float* w        = s2 + T_;                // 1024
  float* c        = w + T_;                 // 1024
  float* wp       = c + T_;                 // 16384
  float* wk       = wp + L_ * D_;           // 16384
  float* wv       = wk + L_ * P_;           // 16384
  float* ep       = wv + L_ * P_;           // 16384
  float* ev       = ep + L_ * P_;           // 16384
  float* eln      = ev + L_ * D_;           // 16384
  float* h        = eln + L_ * D_;          // 65536
  float* mlp      = h + L_ * H_;            // 16384
  float* partial  = bmean;                  // reuse: 32 * 16384 = 524288

  float* out_aligned = (float*)d_out;                       // [L,D]
  float* out_prompts = out_aligned + L_ * D_;               // [T,L,D]
  float* out_tw      = out_prompts + (size_t)T_ * L_ * D_;  // [T]
  float* out_attn    = out_tw + T_;                         // [L,L]

  const float scale = 0.044194173824159216f;  // 1/sqrt(512)

  k_bmean<<<T_, D_ / 4, 0, stream>>>(base, bmean);
  k_taskvec<<<1, P_, 0, stream>>>(temb, Wt, bt, task_vec);
  // key_repr = bmean @ Wk + bk : M=1024,N=512 -> 64*8 = 512 waves
  k_wmma_gemm<0><<<128, 128, 0, stream>>>(bmean, Wk, bk, key_repr, T_, P_, D_);
  // query = new_prompt @ Wq + bq : 2*8 = 16 waves
  k_wmma_gemm<0><<<4, 128, 0, stream>>>(newp, Wq, bq, query, L_, P_, D_);
  k_scores<<<T_, 256, 0, stream>>>(key_repr, task_vec, scores, scale);
  k_softmax<<<1, 1024, 0, stream>>>(scores, w0);
  k_pooledq<<<1, P_, 0, stream>>>(query, pooled_q);
  k_s2<<<T_, 256, 0, stream>>>(key_repr, bk, pooled_q, w0, s2, scale);
  k_combine<<<1, 1024, 0, stream>>>(s2, w0, w, c, out_tw);
  k_wp_partial<<<dim3((L_ * D_ / 4) / 256, 32), 256, 0, stream>>>(base, c, partial);
  k_wp_reduce<<<(L_ * D_ / 4) / 256, 256, 0, stream>>>(partial, wp);
  k_wmma_gemm<0><<<4, 128, 0, stream>>>(wp, Wk, bk, wk, L_, P_, D_);
  k_wmma_gemm<0><<<4, 128, 0, stream>>>(wp, Wv, bv, wv, L_, P_, D_);
  k_feature<<<1, 1024, 0, stream>>>(query, wk, wv, ep, out_attn, scale);
  k_wmma_gemm<0><<<4, 128, 0, stream>>>(ep, Wo, bo, ev, L_, D_, P_);
  k_addln<<<L_, D_, 0, stream>>>(wp, ev, g_in, b_in, eln);
  // h = relu(eln @ Wa1 + ba1) : M=32,N=2048 -> 2*32 = 64 waves
  k_wmma_gemm<2><<<16, 128, 0, stream>>>(eln, Wa1, ba1, h, L_, H_, D_);
  // mlp = h @ Wa2 + ba2 : K=2048
  k_wmma_gemm<0><<<4, 128, 0, stream>>>(h, Wa2, ba2, mlp, L_, D_, H_);
  k_addln<<<L_, D_, 0, stream>>>(eln, mlp, g_out, b_out, out_aligned);
  k_broadcast<<<(T_ * L_ * D_ / 4) / 256, 256, 0, stream>>>((const f32x4*)out_aligned,
                                                            (f32x4*)out_prompts);
}
